// CrossNet_19859928776870
// MI455X (gfx1250) — matrix operand order repaired and checked by reading, not compiled, not run on hardware
//
#include <hip/hip_runtime.h>
#include <hip/hip_bf16.h>
#include <math.h>

typedef __attribute__((ext_vector_type(16))) __bf16 v16bf;
typedef __attribute__((ext_vector_type(8)))  float  v8f;

#define NB   128
#define RR   512
#define CC   81
#define DKK  64
#define RK   10
#define KP   96      // 81 padded to multiple of 32 for bf16 WMMA
#define XSTR 104     // xb_s row stride (bf16): 208B = 13*16, bank-conflict free
#define KSTR 72      // k_s / q_s row stride (bf16): 144B = 9*16, bank-conflict free
#define WSTR 104     // transposed-weight row stride (bf16)
#define SSTR 516     // scores row stride (f32)

struct alignas(16) Smem {
  __bf16 xb_s[RR * XSTR];       // 106496 B  x tile in bf16 (padded cols 81..95 = 0)
  __bf16 k_s[RR * KSTR];        //  73728 B  k in bf16, row-major
  __bf16 wqT[64 * WSTR];        //  13312 B  Wq^T  [n][k]
  __bf16 wkT[64 * WSTR];        //  13312 B  Wk^T
  __bf16 wfcT[KP * WSTR];       //  19968 B  Wfc^T (padded to 96 rows)
  __bf16 q_s[16 * KSTR];        //   2304 B  current q tile
  float  scores[16 * SSTR];     //  33024 B  score tile 16 x 512
  float  prior[CC * CC];        //  26244 B
  int    labels[RR];
  float  xmax[RR];
  float  rowmax[16];
  float  rowden[16];
  float  topv[16][RK];
  int    topj[16][RK];
  float  r_s[16 * KP];          //   6144 B  row-local scatter target
  float  bq_s[64], bk_s[64], bfc_s[KP];
};
static_assert(sizeof(Smem) <= 320 * 1024, "LDS budget");

static __device__ __forceinline__ v8f wmma_bf16(v16bf a, v16bf b, v8f c) {
  return __builtin_amdgcn_wmma_f32_16x16x32_bf16(false, a, false, b, (short)0, c,
                                                 false, false);
}

// A-fragment from row-major bf16 LDS row pointer (already offset by K-block base)
static __device__ __forceinline__ v16bf a_frag_lds(const __bf16* prow, int h) {
  union { v16bf v; uint4 u[2]; } t;
  t.u[0] = *(const uint4*)(prow + 16 * h);
  t.u[1] = *(const uint4*)(prow + 16 * h + 8);
  return t.v;
}

// B-fragment: 16 contiguous bf16 (K=16h..16h+15) from LDS
static __device__ __forceinline__ v16bf b_frag_lds(const __bf16* p) {
  union { v16bf v; uint4 u[2]; } t;
  t.u[0] = ((const uint4*)p)[0];
  t.u[1] = ((const uint4*)p)[1];
  return t.v;
}

// A-fragment from f32 r_s with ReLU fused
static __device__ __forceinline__ v16bf a_frag_relu(const float* prow, int cb, int h) {
  v16bf v;
#pragma unroll
  for (int c = 0; c < 8; ++c) {
    v[c]     = (__bf16)fmaxf(prow[cb + 16 * h + c], 0.f);
    v[c + 8] = (__bf16)fmaxf(prow[cb + 16 * h + 8 + c], 0.f);
  }
  return v;
}

#if __has_builtin(__builtin_amdgcn_global_load_async_to_lds_b128)
#define HAVE_ASYNC_LDS 1
typedef int v4i_vec __attribute__((vector_size(16)));
typedef __attribute__((address_space(1))) v4i_vec* g4p_t;
typedef __attribute__((address_space(3))) v4i_vec* l4p_t;
static __device__ __forceinline__ g4p_t to_g4p(const void* p) {
  return (g4p_t)(uintptr_t)p;                 // inttoptr -> as(1) v4i*
}
static __device__ __forceinline__ l4p_t to_l4p(void* p) {
  // generic LDS address: low 32 bits are the LDS offset (aperture rule)
  return (l4p_t)(unsigned int)(uintptr_t)p;   // inttoptr -> as(3) v4i*
}
#endif

__global__ __launch_bounds__(256)
void crossnet_batch_kernel(const float* __restrict__ x,
                           const float* __restrict__ Wq, const float* __restrict__ bq,
                           const float* __restrict__ Wk, const float* __restrict__ bk,
                           const float* __restrict__ Wfc, const float* __restrict__ bfc,
                           const float* __restrict__ prior,
                           float* __restrict__ out) {
  __shared__ Smem s;
  const int tid  = threadIdx.x;
  const int lane = tid & 31;
  const int wave = tid >> 5;
  const int m16  = lane & 15;   // A-row / B-col / D-col index within tile
  const int h    = lane >> 4;   // lane half
  const int b    = blockIdx.x;
  const float* xb = x + (size_t)b * RR * CC;
  float* outb     = out + (size_t)b * RR * CC;

  // ---- async-copy prior table to LDS (overlaps with rest of staging) ----
#ifdef HAVE_ASYNC_LDS
  for (int i4 = tid * 4; i4 < CC * CC - 1; i4 += 256 * 4)
    __builtin_amdgcn_global_load_async_to_lds_b128(to_g4p(prior + i4),
                                                   to_l4p(&s.prior[i4]), 0, 0);
  if (tid == 0) s.prior[CC * CC - 1] = prior[CC * CC - 1];
#else
  for (int i = tid; i < CC * CC; i += 256) s.prior[i] = prior[i];
#endif

  // ---- stage x (bf16, zero-padded K), weights, biases ----
  for (int i = tid; i < RR * KP; i += 256) {
    int r2 = i / KP, c2 = i % KP;
    s.xb_s[r2 * XSTR + c2] = (c2 < CC) ? (__bf16)xb[r2 * CC + c2] : (__bf16)0.f;
  }
  for (int i = tid; i < 64 * KP; i += 256) {
    int n = i / KP, k2 = i % KP;
    __bf16 zq = (__bf16)0.f, zk = (__bf16)0.f;
    if (k2 < CC) { zq = (__bf16)Wq[k2 * DKK + n]; zk = (__bf16)Wk[k2 * DKK + n]; }
    s.wqT[n * WSTR + k2] = zq;
    s.wkT[n * WSTR + k2] = zk;
  }
  for (int i = tid; i < KP * KP; i += 256) {
    int n = i / KP, k2 = i % KP;
    s.wfcT[n * WSTR + k2] =
        (n < CC && k2 < CC) ? (__bf16)Wfc[k2 * CC + n] : (__bf16)0.f;
  }
  if (tid < 64) { s.bq_s[tid] = bq[tid]; s.bk_s[tid] = bk[tid]; }
  if (tid < KP) s.bfc_s[tid] = (tid < CC) ? bfc[tid] : 0.f;

  // ---- labels (argmax) and row max values (f32 exact) ----
  for (int r2 = tid; r2 < RR; r2 += 256) {
    const float* xr = xb + r2 * CC;
    float mx = xr[0]; int mi = 0;
    for (int c2 = 1; c2 < CC; ++c2) {
      float v = xr[c2];
      if (v > mx) { mx = v; mi = c2; }
    }
    s.labels[r2] = mi;
    s.xmax[r2]   = mx;
  }
#ifdef HAVE_ASYNC_LDS
#if __has_builtin(__builtin_amdgcn_s_wait_asynccnt)
  __builtin_amdgcn_s_wait_asynccnt(0);
#else
  asm volatile("s_wait_asynccnt 0" ::: "memory");
#endif
#endif
  __syncthreads();

  // ---- Phase A: k = xb @ Wk + bk -> bf16 LDS (512x64) ----
  for (int it = 0; it < 4; ++it) {
    int r0 = (wave + 8 * it) * 16;
    const __bf16* xrow = &s.xb_s[(r0 + m16) * XSTR];
    v16bf a0 = a_frag_lds(xrow + 0,  h);
    v16bf a1 = a_frag_lds(xrow + 32, h);
    v16bf a2 = a_frag_lds(xrow + 64, h);
#pragma unroll
    for (int nt = 0; nt < 4; ++nt) {
      int n0 = nt * 16;
      v8f acc = {};
      acc = wmma_bf16(a0, b_frag_lds(&s.wkT[(n0 + m16) * WSTR + 0  + 16 * h]), acc);
      acc = wmma_bf16(a1, b_frag_lds(&s.wkT[(n0 + m16) * WSTR + 32 + 16 * h]), acc);
      acc = wmma_bf16(a2, b_frag_lds(&s.wkT[(n0 + m16) * WSTR + 64 + 16 * h]), acc);
      float bias = s.bk_s[n0 + m16];
      __bf16* kp = &s.k_s[(r0 + 8 * h) * KSTR + n0 + m16];
#pragma unroll
      for (int v = 0; v < 8; ++v) kp[v * KSTR] = (__bf16)(acc[v] + bias);
    }
  }
  __syncthreads();

  // ---- Phase B: per 16-row tile ----
  for (int rt = 0; rt < RR / 16; ++rt) {
    const int i0 = rt * 16;

    // B1: q tile (waves 0..3, one 16-col slab each)
    if (wave < 4) {
      int n0 = wave * 16;
      const __bf16* xrow = &s.xb_s[(i0 + m16) * XSTR];
      v16bf a0 = a_frag_lds(xrow + 0,  h);
      v16bf a1 = a_frag_lds(xrow + 32, h);
      v16bf a2 = a_frag_lds(xrow + 64, h);
      v8f acc = {};
      acc = wmma_bf16(a0, b_frag_lds(&s.wqT[(n0 + m16) * WSTR + 0  + 16 * h]), acc);
      acc = wmma_bf16(a1, b_frag_lds(&s.wqT[(n0 + m16) * WSTR + 32 + 16 * h]), acc);
      acc = wmma_bf16(a2, b_frag_lds(&s.wqT[(n0 + m16) * WSTR + 64 + 16 * h]), acc);
      float bias = s.bq_s[n0 + m16];
      __bf16* qp = &s.q_s[(8 * h) * KSTR + n0 + m16];
#pragma unroll
      for (int v = 0; v < 8; ++v) qp[v * KSTR] = (__bf16)(acc[v] + bias);
    }
    __syncthreads();

    // B2: scores 16x512 = q_tile @ k^T, scaled by 1/sqrt(64)
    {
      v16bf qa0 = a_frag_lds(&s.q_s[m16 * KSTR + 0],  h);
      v16bf qa1 = a_frag_lds(&s.q_s[m16 * KSTR + 32], h);
#pragma unroll
      for (int jj = 0; jj < 4; ++jj) {
        int j0 = (wave * 4 + jj) * 16;
        v8f acc = {};
        acc = wmma_bf16(qa0, b_frag_lds(&s.k_s[(j0 + m16) * KSTR + 0  + 16 * h]), acc);
        acc = wmma_bf16(qa1, b_frag_lds(&s.k_s[(j0 + m16) * KSTR + 32 + 16 * h]), acc);
        float* sp = &s.scores[(8 * h) * SSTR + j0 + m16];
#pragma unroll
        for (int v = 0; v < 8; ++v) sp[v * SSTR] = acc[v] * 0.125f;
      }
    }
    __syncthreads();

    // B3: softmax stats — 16 lanes per row, entirely intra-wave (no barriers)
    const int r = tid >> 4, c = tid & 15;
    float* srow = &s.scores[r * SSTR];
    {
      float m = -3.4e38f;
      for (int k2 = c * 32; k2 < c * 32 + 32; ++k2) m = fmaxf(m, srow[k2]);
      m = fmaxf(m, __shfl_xor(m, 8, 16));
      m = fmaxf(m, __shfl_xor(m, 4, 16));
      m = fmaxf(m, __shfl_xor(m, 2, 16));
      m = fmaxf(m, __shfl_xor(m, 1, 16));
      float sm = 0.f;
      for (int k2 = c * 32; k2 < c * 32 + 32; ++k2) sm += __expf(srow[k2] - m);
      sm += __shfl_xor(sm, 8, 16);
      sm += __shfl_xor(sm, 4, 16);
      sm += __shfl_xor(sm, 2, 16);
      sm += __shfl_xor(sm, 1, 16);
      if (c == 0) { s.rowmax[r] = m; s.rowden[r] = sm; }
    }

    // B4: top-10 by iterative argmax with -inf zap (intra-wave, no barriers)
    for (int t = 0; t < RK; ++t) {
      float bm = -3.4e38f; int bi = 0;
      for (int k2 = c * 32; k2 < c * 32 + 32; ++k2) {
        float v = srow[k2];
        if (v > bm) { bm = v; bi = k2; }
      }
#pragma unroll
      for (int off = 8; off >= 1; off >>= 1) {
        float ov = __shfl_xor(bm, off, 16);
        int   oi = __shfl_xor(bi, off, 16);
        if (ov > bm || (ov == bm && oi < bi)) { bm = ov; bi = oi; }
      }
      if (c == 0) {
        s.topv[r][t] = bm;
        s.topj[r][t] = bi;
        srow[bi] = -3.4e38f;   // same-wave LDS ops are in-order: next scan sees it
      }
    }

    // B5: zero row-local r, then scatter top-k contributions
    for (int k2 = tid; k2 < 16 * KP; k2 += 256) s.r_s[k2] = 0.f;
    __syncthreads();   // also publishes rowmax/rowden/topv/topj across waves
    if (tid < 16) {
      int i  = tid;
      int li = s.labels[i0 + i];
      float rm = s.rowmax[i], rd = s.rowden[i];
      for (int t = 0; t < RK; ++t) {
        int j = s.topj[i][t];
        float val = __expf(s.topv[i][t] - rm) / rd;
        int lj = s.labels[j];
        if (lj != li)
          s.r_s[i * KP + lj] += s.prior[lj * CC + li] * val * s.xmax[j];
      }
    }
    __syncthreads();

    // B7: out = sigmoid(relu(r) @ Wfc + bfc)  (waves 0..5, one 16-col slab each)
    if (wave < 6) {
      int n0 = wave * 16;
      const float* prow = &s.r_s[m16 * KP];
      v16bf a0 = a_frag_relu(prow, 0, h);
      v16bf a1 = a_frag_relu(prow, 32, h);
      v16bf a2 = a_frag_relu(prow, 64, h);
      v8f acc = {};
      acc = wmma_bf16(a0, b_frag_lds(&s.wfcT[(n0 + m16) * WSTR + 0  + 16 * h]), acc);
      acc = wmma_bf16(a1, b_frag_lds(&s.wfcT[(n0 + m16) * WSTR + 32 + 16 * h]), acc);
      acc = wmma_bf16(a2, b_frag_lds(&s.wfcT[(n0 + m16) * WSTR + 64 + 16 * h]), acc);
      int col = n0 + m16;
      float bias = s.bfc_s[col];
      if (col < CC) {
        float* op = &outb[(i0 + 8 * h) * CC + col];
#pragma unroll
        for (int v = 0; v < 8; ++v) {
          float val = acc[v] + bias;
          op[v * CC] = 1.f / (1.f + __expf(-val));
        }
      }
    }
    __syncthreads();
  }
}

extern "C" void kernel_launch(void* const* d_in, const int* in_sizes, int n_in,
                              void* d_out, int out_size, void* d_ws, size_t ws_size,
                              hipStream_t stream) {
  (void)in_sizes; (void)n_in; (void)out_size; (void)d_ws; (void)ws_size;
  const float* x     = (const float*)d_in[0];
  const float* Wq    = (const float*)d_in[1];
  const float* bq    = (const float*)d_in[2];
  const float* Wk    = (const float*)d_in[3];
  const float* bk    = (const float*)d_in[4];
  const float* Wfc   = (const float*)d_in[5];
  const float* bfc   = (const float*)d_in[6];
  const float* prior = (const float*)d_in[7];
  float* out = (float*)d_out;
  crossnet_batch_kernel<<<NB, 256, 0, stream>>>(x, Wq, bq, Wk, bk, Wfc, bfc,
                                                prior, out);
}
